// SASTGNN_38482906972562
// MI455X (gfx1250) — compile-verified, matrix operand order checked by reading, not run on hardware
//
#include <hip/hip_runtime.h>
#include <hip/hip_bf16.h>

// SASTGNN forward for gfx1250 (MI455X). All GEMMs via v_wmma_f32_16x16x32_f16,
// attention softmax fused into register-synthesized WMMA A-fragments.

#define B_  4
#define S_  12
#define P_  3
#define N_  1024
#define F_  64
#define NH_ 4
#define HD_ 16
#define MROWS (B_ * N_)      // 4096 LSTM rows
#define MG    (B_ * P_)      // 12 GAT graphs
#define ALPHA_ 0.2f

typedef _Float16 f16;
typedef __attribute__((ext_vector_type(16))) _Float16 v16h;
typedef __attribute__((ext_vector_type(8)))  float    v8f;

// ---------------- WMMA fragment helpers (ISA 7.12.2 layouts, wave32) --------

__device__ __forceinline__ int frag_k(int slot, int lane) {
  int v = slot >> 1, h = slot & 1;
  int k = (v < 4) ? (v * 2 + h) : (16 + (v - 4) * 2 + h);
  return k + ((lane >> 4) << 3);            // upper half-lanes: +8
}

__device__ __forceinline__ v16h load_frag_a(const f16* src, int ld, int lane) {
  v16h f; int m = lane & 15;
#pragma unroll
  for (int s = 0; s < 16; ++s) f[s] = src[m * ld + frag_k(s, lane)];
  return f;
}

__device__ __forceinline__ v16h load_frag_a_f32(const float* src, int ld, int lane) {
  v16h f; int m = lane & 15;
#pragma unroll
  for (int s = 0; s < 16; ++s) f[s] = (f16)src[m * ld + frag_k(s, lane)];
  return f;
}

// B stored as [n][k] row-major (i.e. transposed weights), ld = k-stride
__device__ __forceinline__ v16h load_frag_b(const f16* src, int ld, int lane) {
  v16h f; int n = lane & 15;
#pragma unroll
  for (int s = 0; s < 16; ++s) f[s] = src[n * ld + frag_k(s, lane)];
  return f;
}

__device__ __forceinline__ v8f wmma16(v16h a, v16h b, v8f c) {
  return __builtin_amdgcn_wmma_f32_16x16x32_f16(false, a, false, b,
                                                (short)0, c, false, false);
}

// v_rcp_f32-based activations (avoid v_div_scale IEEE division sequences)
__device__ __forceinline__ float sigf(float x) {
  return __builtin_amdgcn_rcpf(1.0f + __expf(-x));
}
__device__ __forceinline__ float tanh_fast(float x) {
  return 2.0f * __builtin_amdgcn_rcpf(1.0f + __expf(-2.0f * x)) - 1.0f;
}
__device__ __forceinline__ float leaky(float x) { return x > 0.f ? x : ALPHA_ * x; }
__device__ __forceinline__ float eluf(float x)  { return x > 0.f ? x : (__expf(x) - 1.0f); }

// ---------------- K0: weight repack (f16, transposed / concatenated) --------

__global__ void k_prep(const float* Wih_u, const float* Whh_u,
                       const float* bih_u, const float* bhh_u,
                       const float* Wih_l, const float* Whh_l,
                       const float* bih_l, const float* bhh_l,
                       const float* Wg, const float* Wo, const float* Wrl,
                       f16* WcatU, f16* WcatL, float* BU, float* BL,
                       f16* WgT, f16* WoT, f16* WrlT) {
  int id = blockIdx.x * blockDim.x + threadIdx.x;
  if (id < 256 * 128) {                       // gate weights, K-concat [x|h]
    int j = id >> 7, k = id & 127;
    WcatU[id] = (f16)(k < 64 ? Wih_u[j * 64 + k] : Whh_u[j * 64 + k - 64]);
    WcatL[id] = (f16)(k < 64 ? Wih_l[j * 64 + k] : Whh_l[j * 64 + k - 64]);
    return;
  }
  id -= 256 * 128;
  if (id < 256) { BU[id] = bih_u[id] + bhh_u[id]; BL[id] = bih_l[id] + bhh_l[id]; return; }
  id -= 256;
  if (id < NH_ * HD_ * F_) {                  // WgT[h][k][f] = Wg[h][f][k]
    int h = id / (HD_ * F_), r = id % (HD_ * F_), k = r / F_, f = r % F_;
    WgT[id] = (f16)Wg[(h * F_ + f) * HD_ + k];
    return;
  }
  id -= NH_ * HD_ * F_;
  if (id < F_ * F_) {                         // WoT/WrlT[f][c] = W[c][f]
    int f = id / F_, c = id % F_;
    WoT[id]  = (f16)Wo[c * F_ + f];
    WrlT[id] = (f16)Wrl[c * F_ + f];
  }
}

// ---------------- K1: input projection (K=2) --------------------------------

__global__ void k_proj(const float* __restrict__ x, const float* __restrict__ Wp,
                       const float* __restrict__ bp, float* __restrict__ H0, int total) {
  int id = blockIdx.x * blockDim.x + threadIdx.x;
  if (id >= total) return;
  int f = id & 63, node = id >> 6;
  H0[id] = x[node * 2 + 0] * Wp[f] + x[node * 2 + 1] * Wp[64 + f] + bp[f];
}

// ---------------- K2/K8: persistent-tile LSTM (WMMA gate GEMM) --------------
// Rows m in tiles of 16; state in LDS; 8 waves x 2 gate N-tiles of 16.
// Gate weights (8 B-fragments / wave) are hoisted into registers across all
// timesteps. Input x[m,t,f] = X[(m*T+t)*64+f]; writes OUT = h_t + X.

__global__ void __launch_bounds__(256)
k_lstm(const float* __restrict__ X, const f16* __restrict__ Wcat,
       const float* __restrict__ Bsum, const float* h0, const float* c0,
       int T, float* __restrict__ OUT) {
  __shared__ f16   At[16][128];
  __shared__ float Z[16][256];
  __shared__ float Hs[16][64];
  __shared__ float Cs[16][64];
  int m0 = blockIdx.x * 16;
  int tid = threadIdx.x, lane = tid & 31, wave = tid >> 5;

  // preload this wave's two gate-weight N-tiles (loop-invariant)
  v16h bfr[2][4];
#pragma unroll
  for (int half = 0; half < 2; ++half) {
    int nt = wave + half * 8;
#pragma unroll
    for (int kc = 0; kc < 4; ++kc)
      bfr[half][kc] = load_frag_b(Wcat + (nt * 16) * 128 + kc * 32, 128, lane);
  }
  int colA = wave * 16 + (lane & 15);
  float bs[2] = { Bsum[colA], Bsum[colA + 128] };

  for (int i = tid; i < 16 * 64; i += 256) {
    int r = i >> 6, f = i & 63;
    Hs[r][f] = h0 ? h0[f] : 0.f;
    Cs[r][f] = c0 ? c0[f] : 0.f;
  }
  __syncthreads();

  for (int t = 0; t < T; ++t) {
    for (int i = tid; i < 16 * 128; i += 256) {        // A = [x_t | h_{t-1}]
      int r = i >> 7, k = i & 127;
      float v = (k < 64) ? X[((m0 + r) * T + t) * 64 + k] : Hs[r][k - 64];
      At[r][k] = (f16)v;
    }
    __syncthreads();
    v16h afr[4];
#pragma unroll
    for (int kc = 0; kc < 4; ++kc)
      afr[kc] = load_frag_a(&At[0][0] + kc * 32, 128, lane);
    int g = lane >> 4;
#pragma unroll
    for (int half = 0; half < 2; ++half) {             // 2 gate N-tiles / wave
      v8f acc = {};
#pragma unroll
      for (int kc = 0; kc < 4; ++kc)
        acc = wmma16(afr[kc], bfr[half][kc], acc);
      int col = colA + half * 128;
#pragma unroll
      for (int r = 0; r < 8; ++r) Z[r + g * 8][col] = acc[r] + bs[half];
    }
    __syncthreads();
    for (int i = tid; i < 16 * 64; i += 256) {         // gates i,f,g,o
      int r = i >> 6, f = i & 63;
      float zi = Z[r][f], zf = Z[r][64 + f], zg = Z[r][128 + f], zo = Z[r][192 + f];
      float c = sigf(zf) * Cs[r][f] + sigf(zi) * tanh_fast(zg);
      float h = sigf(zo) * tanh_fast(c);
      Cs[r][f] = c; Hs[r][f] = h;
      int idx = ((m0 + r) * T + t) * 64 + f;
      OUT[idx] = h + X[idx];
    }
    __syncthreads();
  }
}

// ---------------- K3: history conv (12 -> 3 channels) -----------------------

__global__ void k_conv(const float* __restrict__ LO, const float* __restrict__ Wc,
                       const float* __restrict__ bc, float* __restrict__ LOP, int total) {
  int id = blockIdx.x * blockDim.x + threadIdx.x;
  if (id >= total) return;
  int f = id & 63, rest = id >> 6;
  int n = rest & 1023; rest >>= 10;
  int p = rest % P_, b = rest / P_;
  float acc = bc[p];
#pragma unroll
  for (int s = 0; s < S_; ++s)
    acc += Wc[p * S_ + s] * LO[((b * S_ + s) * N_ + n) * F_ + f];
  LOP[id] = acc;
}

// ---------------- K4: GAT head projection + per-node scores -----------------
// grid = MG*NH*64 blocks x 32 threads (1 wave -> one 16-row tile).

__global__ void __launch_bounds__(32)
k_gproj(const float* __restrict__ LOP, const f16* __restrict__ WgT,
        const float* __restrict__ a_src, const float* __restrict__ a_dst,
        f16* __restrict__ HT, float* __restrict__ SS, float* __restrict__ SD) {
  __shared__ float tile[16][16];
  int bid = blockIdx.x;
  int rt = bid & 63, h = (bid >> 6) & 3, m = bid >> 8;
  int lane = threadIdx.x, n0 = rt * 16;
  const float* Arow = LOP + (m * N_ + n0) * F_;
  v8f acc = {};
#pragma unroll
  for (int kc = 0; kc < 2; ++kc) {
    v16h a = load_frag_a_f32(Arow + kc * 32, F_, lane);
    v16h b = load_frag_b(WgT + h * HD_ * F_ + kc * 32, F_, lane);
    acc = wmma16(a, b, acc);
  }
  int col = lane & 15, g = lane >> 4;
#pragma unroll
  for (int r = 0; r < 8; ++r) {
    int row = r + g * 8;
    float v = acc[r];
    tile[row][col] = v;
    HT[((m * NH_ + h) * HD_ + col) * N_ + (n0 + row)] = (f16)v;  // transposed f16
  }
  __syncthreads();
  int row = lane & 15;
  float s = 0.f;
  if (lane < 16) {
#pragma unroll
    for (int k = 0; k < HD_; ++k) s += tile[row][k] * a_src[h * HD_ + k];
    SS[(m * NH_ + h) * N_ + n0 + row] = s;
  } else {
#pragma unroll
    for (int k = 0; k < HD_; ++k) s += tile[row][k] * a_dst[h * HD_ + k];
    SD[(m * NH_ + h) * N_ + n0 + row] = s;
  }
}

// ---------------- K5: masked-softmax attention x h (layer 1, fused) ---------
// Lanes L and L+16 co-own row (L&15): two-pass max/sum over 1024 neighbors,
// then probabilities synthesized straight into WMMA A-fragment slots.

__global__ void __launch_bounds__(32)
k_att1(const unsigned char* __restrict__ adj, const float* __restrict__ SS,
       const float* __restrict__ SD, const f16* __restrict__ HT,
       float* __restrict__ HID) {
  int bid = blockIdx.x;
  int rt = bid & 63, h = (bid >> 6) & 3, m = bid >> 8;
  int lane = threadIdx.x;
  int i = rt * 16 + (lane & 15), half = lane >> 4;
  const float* ssp = SS + (m * NH_ + h) * N_;
  const float* sdp = SD + (m * NH_ + h) * N_;
  float si = ssp[i];
  const unsigned char* arow = adj + i * N_;
  __builtin_prefetch(arow + lane * 32, 0, 0);          // global_prefetch_b8
  __builtin_prefetch(sdp + lane * 32, 0, 0);

  float mx = -3.0e38f;
  for (int j = half; j < N_; j += 2)
    if (arow[j]) mx = fmaxf(mx, leaky(si + sdp[j]));
  mx = fmaxf(mx, __shfl_xor(mx, 16, 32));
  float sum = 0.f;
  for (int j = half; j < N_; j += 2)
    if (arow[j]) sum += __expf(leaky(si + sdp[j]) - mx);
  sum += __shfl_xor(sum, 16, 32);
  float inv = __builtin_amdgcn_rcpf(sum);

  v8f acc = {};
  const f16* Bb = HT + (m * NH_ + h) * HD_ * N_;
  for (int c = 0; c < 32; ++c) {
    int j0 = c * 32;
    v16h a;
#pragma unroll
    for (int s = 0; s < 16; ++s) {
      int j = j0 + frag_k(s, lane);
      a[s] = (f16)(arow[j] ? __expf(leaky(si + sdp[j]) - mx) * inv : 0.f);
    }
    acc = wmma16(a, load_frag_b(Bb + j0, N_, lane), acc);
  }
  int col = lane & 15, g = lane >> 4;
#pragma unroll
  for (int r = 0; r < 8; ++r) {
    int row = r + g * 8;
    HID[(m * N_ + rt * 16 + row) * F_ + h * HD_ + col] = eluf(acc[r]);
  }
}

// ---------------- K6: h2 = hid @ Wo  (+ layer-2 scores) ---------------------

__global__ void __launch_bounds__(32)
k_h2(const float* __restrict__ HID, const f16* __restrict__ WoT,
     const float* __restrict__ ao_src, const float* __restrict__ ao_dst,
     f16* __restrict__ H2T, float* __restrict__ S2S, float* __restrict__ S2D) {
  __shared__ float tile[16][64];
  int bid = blockIdx.x;
  int rt = bid & 63, m = bid >> 6;
  int lane = threadIdx.x, n0 = rt * 16;
  const float* Arow = HID + (m * N_ + n0) * F_;
  v16h a0 = load_frag_a_f32(Arow,      F_, lane);
  v16h a1 = load_frag_a_f32(Arow + 32, F_, lane);
  int col = lane & 15, g = lane >> 4;
#pragma unroll
  for (int nt = 0; nt < 4; ++nt) {
    v8f acc = {};
    acc = wmma16(a0, load_frag_b(WoT + nt * 16 * F_,      F_, lane), acc);
    acc = wmma16(a1, load_frag_b(WoT + nt * 16 * F_ + 32, F_, lane), acc);
#pragma unroll
    for (int r = 0; r < 8; ++r) {
      int row = r + g * 8;
      float v = acc[r];
      tile[row][nt * 16 + col] = v;
      H2T[(m * F_ + nt * 16 + col) * N_ + n0 + row] = (f16)v;
    }
  }
  __syncthreads();
  int row = lane & 15;
  float s = 0.f;
  if (lane < 16) {
#pragma unroll
    for (int f = 0; f < F_; ++f) s += tile[row][f] * ao_src[f];
    S2S[m * N_ + n0 + row] = s;
  } else {
#pragma unroll
    for (int f = 0; f < F_; ++f) s += tile[row][f] * ao_dst[f];
    S2D[m * N_ + n0 + row] = s;
  }
}

// ---------------- K6b: residual path RES = lo @ Wrl + brl -------------------

__global__ void __launch_bounds__(32)
k_res(const float* __restrict__ LOP, const f16* __restrict__ WrlT,
      const float* __restrict__ brl, float* __restrict__ RES) {
  int bid = blockIdx.x;
  int rt = bid & 63, m = bid >> 6;
  int lane = threadIdx.x, n0 = rt * 16;
  const float* Arow = LOP + (m * N_ + n0) * F_;
  v16h a0 = load_frag_a_f32(Arow,      F_, lane);
  v16h a1 = load_frag_a_f32(Arow + 32, F_, lane);
  int col = lane & 15, g = lane >> 4;
#pragma unroll
  for (int nt = 0; nt < 4; ++nt) {
    v8f acc = {};
    acc = wmma16(a0, load_frag_b(WrlT + nt * 16 * F_,      F_, lane), acc);
    acc = wmma16(a1, load_frag_b(WrlT + nt * 16 * F_ + 32, F_, lane), acc);
#pragma unroll
    for (int r = 0; r < 8; ++r)
      RES[(m * N_ + n0 + r + g * 8) * F_ + nt * 16 + col] = acc[r] + brl[nt * 16 + col];
  }
}

// ---------------- K7: masked-softmax attention x h2 (layer 2, fused) --------

__global__ void __launch_bounds__(32)
k_att2(const unsigned char* __restrict__ adj, const float* __restrict__ S2S,
       const float* __restrict__ S2D, const f16* __restrict__ H2T,
       const float* __restrict__ RES, float* __restrict__ G) {
  int bid = blockIdx.x;
  int rt = bid & 63, m = bid >> 6;
  int lane = threadIdx.x;
  int i = rt * 16 + (lane & 15), half = lane >> 4;
  const float* ssp = S2S + m * N_;
  const float* sdp = S2D + m * N_;
  float si = ssp[i];
  const unsigned char* arow = adj + i * N_;
  __builtin_prefetch(arow + lane * 32, 0, 0);          // global_prefetch_b8
  __builtin_prefetch(sdp + lane * 32, 0, 0);

  float mx = -3.0e38f;
  for (int j = half; j < N_; j += 2)
    if (arow[j]) mx = fmaxf(mx, leaky(si + sdp[j]));
  mx = fmaxf(mx, __shfl_xor(mx, 16, 32));
  float sum = 0.f;
  for (int j = half; j < N_; j += 2)
    if (arow[j]) sum += __expf(leaky(si + sdp[j]) - mx);
  sum += __shfl_xor(sum, 16, 32);
  float inv = __builtin_amdgcn_rcpf(sum);

  v8f acc[4] = {};
  for (int c = 0; c < 32; ++c) {
    int j0 = c * 32;
    v16h a;
#pragma unroll
    for (int s = 0; s < 16; ++s) {
      int j = j0 + frag_k(s, lane);
      a[s] = (f16)(arow[j] ? __expf(leaky(si + sdp[j]) - mx) * inv : 0.f);
    }
    const f16* Bb = H2T + m * F_ * N_ + j0;
#pragma unroll
    for (int nt = 0; nt < 4; ++nt)
      acc[nt] = wmma16(a, load_frag_b(Bb + nt * 16 * N_, N_, lane), acc[nt]);
  }
  int col = lane & 15, g = lane >> 4;
#pragma unroll
  for (int nt = 0; nt < 4; ++nt)
#pragma unroll
    for (int r = 0; r < 8; ++r) {
      int idx = (m * N_ + rt * 16 + r + g * 8) * F_ + nt * 16 + col;
      G[idx] = eluf(acc[nt][r]) + RES[idx];
    }
}

// ---------------- K9: output projection (F=64 -> 1) -------------------------

__global__ void k_out(const float* __restrict__ FIN, const float* __restrict__ Wout,
                      const float* __restrict__ bout, float* __restrict__ out, int rows) {
  int r = blockIdx.x * blockDim.x + threadIdx.x;
  if (r >= rows) return;
  float acc = bout[0];
#pragma unroll
  for (int f = 0; f < F_; ++f) acc += FIN[r * F_ + f] * Wout[f];
  out[r] = acc;
}

// ---------------- launch ----------------------------------------------------

extern "C" void kernel_launch(void* const* d_in, const int* in_sizes, int n_in,
                              void* d_out, int out_size, void* d_ws, size_t ws_size,
                              hipStream_t stream) {
  const float* x      = (const float*)d_in[0];
  const unsigned char* adj = (const unsigned char*)d_in[1];   // jax bool -> 1 byte
  const float* Wp     = (const float*)d_in[2];
  const float* bp     = (const float*)d_in[3];
  const float* Wih_u  = (const float*)d_in[4];
  const float* Whh_u  = (const float*)d_in[5];
  const float* bih_u  = (const float*)d_in[6];
  const float* bhh_u  = (const float*)d_in[7];
  const float* Wc     = (const float*)d_in[8];
  const float* bc     = (const float*)d_in[9];
  const float* Wg     = (const float*)d_in[10];
  const float* a_src  = (const float*)d_in[11];
  const float* a_dst  = (const float*)d_in[12];
  const float* Wo     = (const float*)d_in[13];
  const float* ao_src = (const float*)d_in[14];
  const float* ao_dst = (const float*)d_in[15];
  const float* Wrl    = (const float*)d_in[16];
  const float* brl    = (const float*)d_in[17];
  const float* Wih_l  = (const float*)d_in[18];
  const float* Whh_l  = (const float*)d_in[19];
  const float* bih_l  = (const float*)d_in[20];
  const float* bhh_l  = (const float*)d_in[21];
  const float* h0_l   = (const float*)d_in[22];
  const float* c0_l   = (const float*)d_in[23];
  const float* Wout   = (const float*)d_in[24];
  const float* bout   = (const float*)d_in[25];

  char* w = (char*)d_ws;
  auto alloc = [&](size_t bytes) -> char* {
    char* p = w; w += (bytes + 255) & ~size_t(255); return p;
  };
  f16*   WcatU = (f16*)alloc(256 * 128 * sizeof(f16));
  f16*   WcatL = (f16*)alloc(256 * 128 * sizeof(f16));
  float* BU    = (float*)alloc(256 * sizeof(float));
  float* BL    = (float*)alloc(256 * sizeof(float));
  f16*   WgT   = (f16*)alloc(NH_ * HD_ * F_ * sizeof(f16));
  f16*   WoT   = (f16*)alloc(F_ * F_ * sizeof(f16));
  f16*   WrlT  = (f16*)alloc(F_ * F_ * sizeof(f16));
  float* H0    = (float*)alloc((size_t)B_ * S_ * N_ * F_ * sizeof(float));
  float* LO    = (float*)alloc((size_t)B_ * S_ * N_ * F_ * sizeof(float));
  float* LOP   = (float*)alloc((size_t)MG * N_ * F_ * sizeof(float));
  f16*   HT    = (f16*)alloc((size_t)MG * NH_ * HD_ * N_ * sizeof(f16));
  float* SS    = (float*)alloc((size_t)MG * NH_ * N_ * sizeof(float));
  float* SD    = (float*)alloc((size_t)MG * NH_ * N_ * sizeof(float));
  float* HID   = (float*)alloc((size_t)MG * N_ * F_ * sizeof(float));
  f16*   H2T   = (f16*)alloc((size_t)MG * F_ * N_ * sizeof(f16));
  float* S2S   = (float*)alloc((size_t)MG * N_ * sizeof(float));
  float* S2D   = (float*)alloc((size_t)MG * N_ * sizeof(float));
  float* RES   = (float*)alloc((size_t)MG * N_ * F_ * sizeof(float));
  float* Gbuf  = (float*)alloc((size_t)MG * N_ * F_ * sizeof(float));
  float* FIN   = (float*)alloc((size_t)MG * N_ * F_ * sizeof(float));

  const int prepN = 256 * 128 + 256 + NH_ * HD_ * F_ + F_ * F_;
  k_prep<<<(prepN + 255) / 256, 256, 0, stream>>>(
      Wih_u, Whh_u, bih_u, bhh_u, Wih_l, Whh_l, bih_l, bhh_l,
      Wg, Wo, Wrl, WcatU, WcatL, BU, BL, WgT, WoT, WrlT);

  const int projN = B_ * S_ * N_ * F_;
  k_proj<<<(projN + 255) / 256, 256, 0, stream>>>(x, Wp, bp, H0, projN);

  k_lstm<<<MROWS / 16, 256, 0, stream>>>(H0, WcatU, BU, nullptr, nullptr, S_, LO);

  const int convN = B_ * P_ * N_ * F_;
  k_conv<<<(convN + 255) / 256, 256, 0, stream>>>(LO, Wc, bc, LOP, convN);

  k_gproj<<<MG * NH_ * (N_ / 16), 32, 0, stream>>>(LOP, WgT, a_src, a_dst, HT, SS, SD);
  k_att1 <<<MG * NH_ * (N_ / 16), 32, 0, stream>>>(adj, SS, SD, HT, HID);
  k_h2   <<<MG * (N_ / 16), 32, 0, stream>>>(HID, WoT, ao_src, ao_dst, H2T, S2S, S2D);
  k_res  <<<MG * (N_ / 16), 32, 0, stream>>>(LOP, WrlT, brl, RES);
  k_att2 <<<MG * (N_ / 16), 32, 0, stream>>>(adj, S2S, S2D, H2T, RES, Gbuf);

  k_lstm<<<MROWS / 16, 256, 0, stream>>>(Gbuf, WcatL, BL, h0_l, c0_l, P_, FIN);

  const int rows = B_ * P_ * N_;
  k_out<<<(rows + 255) / 256, 256, 0, stream>>>(FIN, Wout, bout, (float*)d_out, rows);
}
